// MultiheadAttentionWithRelPositionalEncoding_32057635898180
// MI455X (gfx1250) — matrix operand order repaired and checked by reading, not compiled, not run on hardware
//
#include <hip/hip_runtime.h>
#include <hip/hip_bf16.h>
#include <stdint.h>

// ---------------------------------------------------------------------------
// CDNA5 (gfx1250) fused rel-pos multihead attention.
// wave32, WMMA f32_16x16x32_f16 throughout. Flash-style attention with a
// rolling LDS-buffered "band GEMM" implementing the rel-shift, and TDM
// (tensor_load_to_lds) double-buffered staging of K/V tiles into LDS.
// ---------------------------------------------------------------------------

typedef __attribute__((ext_vector_type(16))) _Float16 v16h;
typedef __attribute__((ext_vector_type(8)))  float    v8f;

union HFrag { v16h v; _Float16 h[16]; };

__device__ __forceinline__ int imin_(int a, int b) { return a < b ? a : b; }

__device__ __forceinline__ v8f wmma_acc(v16h a, v16h b, v8f c) {
  // D = A(16x32 f16) * B(32x16 f16) + C(16x16 f32)
  return __builtin_amdgcn_wmma_f32_16x16x32_f16(
      false, a, false, b, (short)0, c, false, false);
}

// A-fragment: rows row0..row0+15 of row-major f16 matrix, K-chunk kc..kc+31.
__device__ __forceinline__ v16h load_a_f16(const _Float16* __restrict__ X,
                                           int ld, int row0, int maxRow,
                                           int kc, int lane) {
  int r  = imin_(row0 + (lane & 15), maxRow);
  int ko = kc + ((lane >> 4) << 3);
  const _Float16* p = X + (size_t)r * ld + ko;
  HFrag f;
#pragma unroll
  for (int i = 0; i < 8; ++i) f.h[i] = p[i];
#pragma unroll
  for (int i = 0; i < 8; ++i) f.h[8 + i] = p[16 + i];
  return f.v;
}

// Same, converting an f32 source on the fly.
__device__ __forceinline__ v16h load_a_f32(const float* __restrict__ X,
                                           int ld, int row0, int maxRow,
                                           int kc, int lane) {
  int r  = imin_(row0 + (lane & 15), maxRow);
  int ko = kc + ((lane >> 4) << 3);
  const float* p = X + (size_t)r * ld + ko;
  HFrag f;
#pragma unroll
  for (int i = 0; i < 8; ++i) f.h[i] = (_Float16)p[i];
#pragma unroll
  for (int i = 0; i < 8; ++i) f.h[8 + i] = (_Float16)p[16 + i];
  return f.v;
}

// B-fragment from global: column (col0+n) stored contiguously (ld elems).
__device__ __forceinline__ v16h load_b_f16(const _Float16* __restrict__ X,
                                           int ld, int col0, int maxCol,
                                           int kc, int lane) {
  int c = imin_(col0 + (lane & 15), maxCol);
  const _Float16* p = X + (size_t)c * ld + kc + ((lane >> 4) << 3);
  HFrag f;
#pragma unroll
  for (int i = 0; i < 8; ++i) f.h[i] = p[i];
#pragma unroll
  for (int i = 0; i < 8; ++i) f.h[8 + i] = p[16 + i];
  return f.v;
}

// B-fragment from LDS-staged tile (no clamping).
__device__ __forceinline__ v16h load_b_lds(const _Float16* p0, int ld,
                                           int col0, int kc, int lane) {
  const _Float16* p =
      p0 + (size_t)(col0 + (lane & 15)) * ld + kc + ((lane >> 4) << 3);
  HFrag f;
#pragma unroll
  for (int i = 0; i < 8; ++i) f.h[i] = p[i];
#pragma unroll
  for (int i = 0; i < 8; ++i) f.h[8 + i] = p[16 + i];
  return f.v;
}

// ---------------------------------------------------------------------------
// Tensor Data Mover wrapper (D# per CDNA5 ISA ch.8): 1D/2D f16 tile -> LDS
// with optional LDS row padding.  data_size=2 (4-byte units).
// ---------------------------------------------------------------------------
#if __has_builtin(__builtin_amdgcn_tensor_load_to_lds)
#define HAVE_TDM 1
typedef __attribute__((ext_vector_type(4))) unsigned int tdm_u32x4;
typedef __attribute__((ext_vector_type(8))) int          tdm_i32x8;
typedef __attribute__((ext_vector_type(4))) int          tdm_i32x4;

__device__ __forceinline__ void tdm_load_tile(
    uint32_t ldsAddr, const void* gptr, unsigned tileD0dw, unsigned tileD1,
    unsigned strideDw, unsigned padInterval, unsigned padAmount) {
  uint64_t ga = (uint64_t)(uintptr_t)gptr;
  tdm_u32x4 g0;
  g0.x = 1u;                                                  // count=1
  g0.y = ldsAddr;                                             // lds_addr
  g0.z = (uint32_t)ga;                                        // global_addr lo
  g0.w = ((uint32_t)(ga >> 32) & 0x01FFFFFFu) | (2u << 30);   // hi | type=2
  tdm_i32x8 g1;
  g1[0] = (int)((2u << 16) | (1u << 20) |                     // dsz=4B, pad_en
                (padInterval << 22) | (padAmount << 25));
  g1[1] = (int)0xFFFF0000u;              // tensor_dim0 = 0x7FFFFFFF (lo16<<16)
  g1[2] = (int)0xFFFF7FFFu;              // dim0 hi16 | tensor_dim1 lo16
  g1[3] = (int)(0x7FFFu | (tileD0dw << 16));  // dim1 hi16 | tile_dim0
  g1[4] = (int)(tileD1 & 0xFFFFu);            // tile_dim1 | tile_dim2=0
  g1[5] = (int)strideDw;                      // tensor_dim0_stride lo32
  g1[6] = 0;
  g1[7] = 0;
  tdm_i32x4 gz = {0, 0, 0, 0};
#if __clang_major__ >= 23
  tdm_i32x8 gz8 = {};
  __builtin_amdgcn_tensor_load_to_lds(g0, g1, gz, gz, gz8, 0);
#else
  __builtin_amdgcn_tensor_load_to_lds(g0, g1, gz, gz, 0);
#endif
}
#endif

// ---------------------------------------------------------------------------
// Kernel 1: f32 -> f16 conversion of the five 512x512 weight matrices.
// ---------------------------------------------------------------------------
__global__ void cvt_weights(const float* __restrict__ wq,
                            const float* __restrict__ wk,
                            const float* __restrict__ wv,
                            const float* __restrict__ wo,
                            const float* __restrict__ wp,
                            _Float16* __restrict__ o) {
  const int N = 512 * 512;
  for (int i = blockIdx.x * blockDim.x + threadIdx.x; i < N;
       i += gridDim.x * blockDim.x) {
    o[i]         = (_Float16)wq[i];
    o[N + i]     = (_Float16)wk[i];
    o[2 * N + i] = (_Float16)wv[i];
    o[3 * N + i] = (_Float16)wo[i];
    o[4 * N + i] = (_Float16)wp[i];
  }
}

// ---------------------------------------------------------------------------
// Kernel 2: WMMA GEMM  out = X(Mx512) @ W^T + bias, templated on MODE and
// A-source type so the K-loop is branch-free.
//  mode 0: Q -> out0=(q+pos_u), out1=(q+pos_v) f16 [b,h,s,d]
//  mode 1: K -> out0 f16 [b,h,s,d]
//  mode 2: V -> out0 f16 [b,h,d,s] (transposed)
//  mode 3: P -> out0 f16 [h, 2S-1, d] (no bias)
//  mode 4: out proj -> outf f32 (final result)
// ---------------------------------------------------------------------------
template <int MODE, bool XH>
__global__ __launch_bounds__(256) void gemm_proj(
    const float* __restrict__ Xf, const _Float16* __restrict__ Xh,
    const _Float16* __restrict__ W, const float* __restrict__ bias,
    int M,
    _Float16* __restrict__ out0, _Float16* __restrict__ out1,
    float* __restrict__ outf,
    const float* __restrict__ pos_u, const float* __restrict__ pos_v) {
  const int Kdim = 512;
  int lane = threadIdx.x & 31;
  int w    = threadIdx.x >> 5;
  long tile = (long)blockIdx.x * 8 + w;
  int tilesM = (M + 15) >> 4;
  if (tile >= (long)tilesM * 32) return;
  int tm = (int)(tile >> 5);
  int tn = (int)(tile & 31);
  int row0 = tm << 4, col0 = tn << 4;
  int maxRow = M - 1;

  v8f acc = {};
#pragma unroll 4
  for (int kc = 0; kc < Kdim; kc += 32) {
    v16h a = XH ? load_a_f16(Xh, Kdim, row0, maxRow, kc, lane)
                : load_a_f32(Xf, Kdim, row0, maxRow, kc, lane);
    v16h bm = load_b_f16(W, Kdim, col0, 511, kc, lane);
    acc = wmma_acc(a, bm, acc);
  }

  int n  = col0 + (lane & 15);
  float bn = (MODE == 3) ? 0.0f : bias[n];
  int hh = n >> 6, d = n & 63;
  float uu = 0.f, vv = 0.f;
  if (MODE == 0) { uu = pos_u[hh * 64 + d]; vv = pos_v[hh * 64 + d]; }

#pragma unroll
  for (int r = 0; r < 8; ++r) {
    int m = row0 + r + ((lane >> 4) << 3);   // C-layout row
    if (m >= M) continue;
    float val = acc[r] + bn;
    if (MODE == 0) {
      int bb = m >> 11, s = m & 2047;
      size_t off = (((size_t)(bb * 8 + hh) * 2048) + s) * 64 + d;
      out0[off] = (_Float16)(val + uu);
      out1[off] = (_Float16)(val + vv);
    } else if (MODE == 1) {
      int bb = m >> 11, s = m & 2047;
      size_t off = (((size_t)(bb * 8 + hh) * 2048) + s) * 64 + d;
      out0[off] = (_Float16)val;
    } else if (MODE == 2) {
      int bb = m >> 11, s = m & 2047;
      size_t off = (((size_t)(bb * 8 + hh) * 64) + d) * 2048 + s;
      out0[off] = (_Float16)val;
    } else if (MODE == 3) {
      size_t off = (((size_t)hh * 4095) + m) * 64 + d;
      out0[off] = (_Float16)val;
    } else {
      outf[(size_t)m * 512 + n] = val;
    }
  }
}

// ---------------------------------------------------------------------------
// Kernel 3: fused flash attention with rel-pos band GEMM.
// Block = 4 waves = 4 consecutive 16-row q-tiles of one (b,h).  Per 32-key
// step the block stages K (32x64, contiguous) and Vt (64x32, strided) tiles
// into padded LDS buffers via TDM (double buffered), then each wave does:
//   2 WMMA content scores + 2 WMMA band-GEMM (rolling rel-shift) + softmax
//   + 4 WMMA PV.
// ---------------------------------------------------------------------------
#define KLD 72  // K tile LDS row stride (halfs): 64 + 8 pad  (36 dwords)
#define VLD 40  // V tile LDS row stride (halfs): 32 + 8 pad  (20 dwords)

__global__ __launch_bounds__(128) void attn_kernel(
    const _Float16* __restrict__ qu16, const _Float16* __restrict__ qv16,
    const _Float16* __restrict__ k16,  const _Float16* __restrict__ vt16,
    const _Float16* __restrict__ p16,  const uint8_t* __restrict__ kmask,
    _Float16* __restrict__ attn16) {
  const int S = 2048, DH = 64, PR = 2 * S - 1;   // PR = 4095
  int lane = threadIdx.x & 31;
  int w    = threadIdx.x >> 5;
  int gw   = blockIdx.x * 4 + w;
  int qt   = gw & 127;          // 4 consecutive q-tiles per block
  int bh   = gw >> 7;           // same (b,h) for whole block
  int b    = bh >> 3, h = bh & 7;
  int q0   = qt << 4;

  __shared__ _Float16 kbuf[2][32 * KLD];   // staged K tiles
  __shared__ _Float16 vbuf[2][64 * VLD];   // staged Vt tiles
  __shared__ float    gl[4][16 * 33];      // rolling band-GEMM tiles
  __shared__ _Float16 pl[4][16 * 40];      // probs relayout C->A
  float*    gld = gl[w];
  _Float16* pld = pl[w];

  const _Float16* Qu = qu16 + (size_t)bh * S * DH;
  const _Float16* Qv = qv16 + (size_t)bh * S * DH;
  const _Float16* Km = k16  + (size_t)bh * S * DH;
  const _Float16* Vt = vt16 + (size_t)bh * DH * S;
  const _Float16* P  = p16  + (size_t)h * PR * DH;
  const uint8_t*  Mk = kmask + (size_t)b * S;

#ifdef HAVE_TDM
  uint32_t kBase = (uint32_t)(uintptr_t)&kbuf[0][0];
  uint32_t vBase = (uint32_t)(uintptr_t)&vbuf[0][0];
#endif

  // Stage K/V tile t (32 keys) into buffer t&1.
  auto stage = [&](int t) {
    int kk  = t << 5;
    int buf = t & 1;
#ifdef HAVE_TDM
    if (w == 0) {
      // K: 1024 contiguous dwords, pad 4 dwords every 32 -> row stride 36 dw
      tdm_load_tile(kBase + (uint32_t)buf * (32 * KLD * 2),
                    Km + (size_t)kk * DH, 1024, 0, 1024, 4, 3);
      // Vt: 64 rows x 16 dwords, global row stride 1024 dw, pad 4 every 16
      tdm_load_tile(vBase + (uint32_t)buf * (64 * VLD * 2),
                    Vt + kk, 16, 64, 1024, 3, 3);
    }
#else
    uint32_t* kw = (uint32_t*)&kbuf[buf][0];
    uint32_t* vw = (uint32_t*)&vbuf[buf][0];
    const uint32_t* gk = (const uint32_t*)(Km + (size_t)kk * DH);
    for (int dw = threadIdx.x; dw < 1024; dw += 128)
      kw[(dw >> 5) * 36 + (dw & 31)] = gk[dw];
    for (int dw = threadIdx.x; dw < 1024; dw += 128) {
      int row = dw >> 4, col = dw & 15;
      vw[row * 20 + col] =
          ((const uint32_t*)(Vt + kk + (size_t)row * S))[col];
    }
#endif
  };

  // Q fragments (16 rows x 64 dh), loaded once.
  v16h qu0 = load_a_f16(Qu, DH, q0, S - 1, 0,  lane);
  v16h qu1 = load_a_f16(Qu, DH, q0, S - 1, 32, lane);
  v16h qv0 = load_a_f16(Qv, DH, q0, S - 1, 0,  lane);
  v16h qv1 = load_a_f16(Qv, DH, q0, S - 1, 32, lane);

  float mrow[8], lrow[8];
#pragma unroll
  for (int r = 0; r < 8; ++r) { mrow[r] = -1.0e30f; lrow[r] = 0.0f; }
  v8f o0 = {}, o1 = {}, o2 = {}, o3 = {};

  const int iRow = ((lane >> 4) << 3);   // +r -> C-layout row
  const int jCol = lane & 15;            // C-layout col

  // Prime band-GEMM slot 0: P rows (S-16-q0)..+15 (always >= 0).
  {
    int pr0 = S - 16 - q0;
    v8f g = {};
    g = wmma_acc(qv0, load_b_f16(P, DH, pr0, PR - 1, 0,  lane), g);
    g = wmma_acc(qv1, load_b_f16(P, DH, pr0, PR - 1, 32, lane), g);
#pragma unroll
    for (int r = 0; r < 8; ++r) gld[(iRow + r) * 33 + jCol] = g[r];
  }

  stage(0);

  const float inv_scale = 0.125f;        // 1/sqrt(64)

  for (int ts2 = 0; ts2 < S / 32; ++ts2) {
    int cur = ts2 & 1;
#ifdef HAVE_TDM
    if (w == 0) __builtin_amdgcn_s_wait_tensorcnt(0);
#endif
    __syncthreads();                     // publish staged tile ts2
    if (ts2 + 1 < S / 32) stage(ts2 + 1);

    const _Float16* kcur = &kbuf[cur][0];
    const _Float16* vcur = &vbuf[cur][0];

    v8f sc[2];
#pragma unroll
    for (int u = 0; u < 2; ++u) {
      int ts = ts2 * 2 + u;
      int kk = (ts2 << 5) + (u << 4);
      // content scores: (Q+u) . K^T over dh=64 (K from LDS)
      v8f ac = {};
      ac = wmma_acc(qu0, load_b_lds(kcur, KLD, u << 4, 0,  lane), ac);
      ac = wmma_acc(qu1, load_b_lds(kcur, KLD, u << 4, 32, lane), ac);
      // new band-GEMM tile: P rows (S + 16*ts - q0)..+15 -> slot (ts+1)&1
      int pr0 = S + (ts << 4) - q0;
      v8f g = {};
      g = wmma_acc(qv0, load_b_f16(P, DH, pr0, PR - 1, 0,  lane), g);
      g = wmma_acc(qv1, load_b_f16(P, DH, pr0, PR - 1, 32, lane), g);
      int slNew = (ts + 1) & 1;
#pragma unroll
      for (int r = 0; r < 8; ++r)
        gld[(iRow + r) * 33 + (slNew << 4) + jCol] = g[r];
      asm volatile("s_wait_dscnt 0x0" ::: "memory");

      float madd = Mk[kk + jCol] ? -1.0e9f : 0.0f;
#pragma unroll
      for (int r = 0; r < 8; ++r) {
        int i = iRow + r;
        int c = jCol - i + 15;             // band offset, 0..30
        int sl = (ts + (c >> 4)) & 1;      // which rolling slot holds it
        float bd = gld[i * 33 + (sl << 4) + (c & 15)];
        sc[u][r] = (ac[r] + bd) * inv_scale + madd;
      }
    }

    // Online softmax across these 32 columns (16-lane half reductions).
#pragma unroll
    for (int r = 0; r < 8; ++r) {
      float mx = fmaxf(sc[0][r], sc[1][r]);
#pragma unroll
      for (int dd = 1; dd < 16; dd <<= 1)
        mx = fmaxf(mx, __shfl_xor(mx, dd, 32));
      float mnew    = fmaxf(mrow[r], mx);
      float rescale = __expf(mrow[r] - mnew);
      float p0 = __expf(sc[0][r] - mnew);
      float p1 = __expf(sc[1][r] - mnew);
      float rs = p0 + p1;
#pragma unroll
      for (int dd = 1; dd < 16; dd <<= 1) rs += __shfl_xor(rs, dd, 32);
      lrow[r] = lrow[r] * rescale + rs;
      mrow[r] = mnew;
      o0[r] *= rescale; o1[r] *= rescale; o2[r] *= rescale; o3[r] *= rescale;
      int i = iRow + r;
      pld[i * 40 + jCol]      = (_Float16)p0;
      pld[i * 40 + 16 + jCol] = (_Float16)p1;
    }
    asm volatile("s_wait_dscnt 0x0" ::: "memory");

    // Re-layout probs (C-layout) into an A-fragment over the 32 keys.
    HFrag pa;
    {
      int rr = lane & 15;
      int ko = ((lane >> 4) << 3);
#pragma unroll
      for (int i2 = 0; i2 < 8; ++i2) pa.h[i2]     = pld[rr * 40 + ko + i2];
#pragma unroll
      for (int i2 = 0; i2 < 8; ++i2) pa.h[8 + i2] = pld[rr * 40 + 16 + ko + i2];
    }

    // O += P(16x32) x V(32x64); Vt tile from LDS (columns contiguous).
    o0 = wmma_acc(pa.v, load_b_lds(vcur, VLD, 0,  0, lane), o0);
    o1 = wmma_acc(pa.v, load_b_lds(vcur, VLD, 16, 0, lane), o1);
    o2 = wmma_acc(pa.v, load_b_lds(vcur, VLD, 32, 0, lane), o2);
    o3 = wmma_acc(pa.v, load_b_lds(vcur, VLD, 48, 0, lane), o3);

    // Prefetch upcoming P rows (per-wave stream).
    if (ts2 + 1 < S / 32)
      __builtin_prefetch(P + (size_t)(S + (ts2 * 2 + 2) * 16 - q0) * DH, 0, 1);
  }

  // Epilogue: normalize and merge heads -> f16 [b, s, h*64+d]
#pragma unroll
  for (int r = 0; r < 8; ++r) {
    int i = iRow + r;
    float inv = 1.0f / lrow[r];
    size_t base = ((size_t)b * S + (q0 + i)) * 512 + h * 64 + jCol;
    attn16[base]      = (_Float16)(o0[r] * inv);
    attn16[base + 16] = (_Float16)(o1[r] * inv);
    attn16[base + 32] = (_Float16)(o2[r] * inv);
    attn16[base + 48] = (_Float16)(o3[r] * inv);
  }
}

// ---------------------------------------------------------------------------
extern "C" void kernel_launch(void* const* d_in, const int* in_sizes, int n_in,
                              void* d_out, int out_size, void* d_ws,
                              size_t ws_size, hipStream_t stream) {
  (void)in_sizes; (void)n_in; (void)out_size; (void)ws_size;
  const float*   query   = (const float*)d_in[0];
  const float*   key     = (const float*)d_in[1];
  const float*   value   = (const float*)d_in[2];
  const uint8_t* kmask   = (const uint8_t*)d_in[3];
  const float*   pos_emb = (const float*)d_in[4];
  // d_in[5] = is_causal (always 0 per reference assert)
  const float* wq_w = (const float*)d_in[6];
  const float* wq_b = (const float*)d_in[7];
  const float* wk_w = (const float*)d_in[8];
  const float* wk_b = (const float*)d_in[9];
  const float* wv_w = (const float*)d_in[10];
  const float* wv_b = (const float*)d_in[11];
  const float* wo_w = (const float*)d_in[12];
  const float* wo_b = (const float*)d_in[13];
  const float* pos_w = (const float*)d_in[14];
  const float* pos_u = (const float*)d_in[15];
  const float* pos_v = (const float*)d_in[16];

  uint8_t* ws = (uint8_t*)d_ws;
  size_t off = 0;
  auto alloc = [&](size_t bytes) -> void* {
    void* p = ws + off;
    off += (bytes + 255) & ~(size_t)255;
    return p;
  };
  const size_t WN  = 512 * 512;
  const size_t BSH = (size_t)4 * 2048 * 512;
  _Float16* w16    = (_Float16*)alloc(5 * WN * sizeof(_Float16));
  _Float16* qu16   = (_Float16*)alloc(BSH * sizeof(_Float16));
  _Float16* qv16   = (_Float16*)alloc(BSH * sizeof(_Float16));
  _Float16* k16    = (_Float16*)alloc(BSH * sizeof(_Float16));
  _Float16* vt16   = (_Float16*)alloc(BSH * sizeof(_Float16));
  _Float16* p16    = (_Float16*)alloc((size_t)8 * 4095 * 64 * sizeof(_Float16));
  _Float16* attn16 = (_Float16*)alloc(BSH * sizeof(_Float16));

  cvt_weights<<<512, 256, 0, stream>>>(wq_w, wk_w, wv_w, wo_w, pos_w, w16);

  gemm_proj<0, false><<<2048, 256, 0, stream>>>(query, nullptr,
      w16 + 0 * WN, wq_b, 8192, qu16, qv16, nullptr, pos_u, pos_v);
  gemm_proj<1, false><<<2048, 256, 0, stream>>>(key, nullptr,
      w16 + 1 * WN, wk_b, 8192, k16, nullptr, nullptr, nullptr, nullptr);
  gemm_proj<2, false><<<2048, 256, 0, stream>>>(value, nullptr,
      w16 + 2 * WN, wv_b, 8192, vt16, nullptr, nullptr, nullptr, nullptr);
  gemm_proj<3, false><<<1024, 256, 0, stream>>>(pos_emb, nullptr,
      w16 + 4 * WN, nullptr, 4095, p16, nullptr, nullptr, nullptr, nullptr);

  attn_kernel<<<1024, 128, 0, stream>>>(qu16, qv16, k16, vt16, p16, kmask,
                                        attn16);

  gemm_proj<4, true><<<2048, 256, 0, stream>>>(nullptr, attn16,
      w16 + 3 * WN, wo_b, 8192, nullptr, nullptr, (float*)d_out,
      nullptr, nullptr);
}
